// SimplifiedPairformerBlock_9775345565947
// MI455X (gfx1250) — compile-verified, matrix operand order checked by reading, not run on hardware
//
#include <hip/hip_runtime.h>

// ---------------------------------------------------------------------------
// SimplifiedPairformerBlock on MI455X (gfx1250, wave32, WMMA bf16,
// async global->LDS staging, DPP16 softmax reductions). N=256, C=128, H=4, d=32
// ---------------------------------------------------------------------------

typedef __bf16 bf16;
typedef __attribute__((ext_vector_type(16))) __bf16 v16bf;
typedef __attribute__((ext_vector_type(8)))  __bf16 v8bf;
typedef __attribute__((ext_vector_type(8)))  float  v8f;

#define NN_TOK 65536   // N*N
#define NSEQ   256
#define CH     128
#define HEADS  4
#define DHEAD  32
#define LN_EPS 1e-5f
#define QK_SC  0.17677669529663687f  // 1/sqrt(32)

// --- LDS byte-offset of a shared-memory object ------------------------------
__device__ __forceinline__ unsigned int lds_off_of(const void* p) {
  return (unsigned int)(unsigned long long)(__attribute__((address_space(3))) const void*)p;
}

// --- gfx1250 async copy: 16B global -> LDS, tracked by ASYNCcnt -------------
__device__ __forceinline__ void async_copy16(unsigned int lds_byte_off,
                                             const bf16* gptr) {
  asm volatile("global_load_async_to_lds_b128 %0, %1, off"
               :
               : "v"(lds_byte_off), "v"(gptr)
               : "memory");
}
__device__ __forceinline__ void wait_async0() {
  asm volatile("s_wait_asynccnt 0x0" ::: "memory");
}

// --- 16-lane max reduction on the VALU DPP16 pipe (rows of 16 = WMMA halves)
// steps: quad_perm(1,0,3,2), quad_perm(2,3,0,1), row_ror:4, row_ror:8
__device__ __forceinline__ float row_max16(float x) {
  int v;
  v = __builtin_amdgcn_update_dpp(__float_as_int(x), __float_as_int(x), 0xB1, 0xF, 0xF, true);
  x = fmaxf(x, __int_as_float(v));
  v = __builtin_amdgcn_update_dpp(__float_as_int(x), __float_as_int(x), 0x4E, 0xF, 0xF, true);
  x = fmaxf(x, __int_as_float(v));
  v = __builtin_amdgcn_update_dpp(__float_as_int(x), __float_as_int(x), 0x124, 0xF, 0xF, true);
  x = fmaxf(x, __int_as_float(v));
  v = __builtin_amdgcn_update_dpp(__float_as_int(x), __float_as_int(x), 0x128, 0xF, 0xF, true);
  x = fmaxf(x, __int_as_float(v));
  return x;
}

// --- 16-bit A/B fragment loader, exact CDNA5 16x32 layout -------------------
// lane m = lane&15 owns one row; lanes 0-15 take K = {0..7,16..23},
// lanes 16-31 take K = {8..15,24..31}; pairs packed low-to-high.
__device__ __forceinline__ v16bf load_frag(const bf16* rowptr, int lane) {
  const int kh = (lane >> 4) << 3;
  v8bf a = *(const v8bf*)(rowptr + kh);
  v8bf b = *(const v8bf*)(rowptr + 16 + kh);
  v16bf f;
#pragma unroll
  for (int i = 0; i < 8; ++i) { f[i] = a[i]; f[i + 8] = b[i]; }
  return f;
}

__device__ __forceinline__ v8f wmma_bf16(v16bf a, v16bf b, v8f c) {
  return __builtin_amdgcn_wmma_f32_16x16x32_bf16(false, a, false, b, (short)0, c,
                                                 false, false);
}

// --- weight transpose + f32->bf16 convert: in[K x Nn] -> out[Nn x K] --------
__global__ void k_tconv(const float* __restrict__ in, bf16* __restrict__ out,
                        int K, int Nn) {
  int t = blockIdx.x * 256 + threadIdx.x;
  if (t < K * Nn) {
    int k = t / Nn, n = t % Nn;
    out[(size_t)n * K + k] = (bf16)in[t];
  }
}

__global__ void k_conv(const float* __restrict__ in, bf16* __restrict__ out, int n) {
  int t = blockIdx.x * 256 + threadIdx.x;
  if (t < n) out[t] = (bf16)in[t];
}

// --- attention additive bias: bias[j,k] = mask? -1e30 : dscale*dist[j,k] ----
__global__ void k_bias(const float* __restrict__ dist,
                       const unsigned char* __restrict__ mask,
                       const float* __restrict__ dsc, float* __restrict__ out,
                       int T) {
  int t = blockIdx.x * 256 + threadIdx.x;  // 65536 total
  int j = t >> 8, k = t & 255;
  unsigned char mb = T ? mask[k * NSEQ + j] : mask[j * NSEQ + k];
  out[t] = mb ? -1e30f : dsc[0] * dist[t];
}

// --- tri-update prefix sums -------------------------------------------------
__global__ __launch_bounds__(128) void k_prefix_row(const float* __restrict__ p,
                                                    float* __restrict__ s) {
  int i = blockIdx.x, c = threadIdx.x;
  float acc = 0.f;
  for (int k = 0; k < i; ++k) acc += p[((size_t)i * NSEQ + k) * CH + c];
  s[i * CH + c] = acc;
}
__global__ __launch_bounds__(128) void k_prefix_col(const float* __restrict__ p,
                                                    float* __restrict__ s) {
  int j = blockIdx.x, c = threadIdx.x;
  float acc = 0.f;
  for (int k = 0; k < j; ++k) acc += p[((size_t)k * NSEQ + j) * CH + c];
  s[j * CH + c] = acc;
}

// --- fused (base + u[i] + w[j]) -> LayerNorm -> f32 + bf16 ------------------
__global__ __launch_bounds__(128) void k_add_ln(
    const float* __restrict__ base, const float* __restrict__ u,
    const float* __restrict__ w, const float* __restrict__ g,
    const float* __restrict__ b, float* __restrict__ dstF,
    bf16* __restrict__ dstB) {
  int idx = blockIdx.x;  // i*N + j
  int i = idx >> 8, j = idx & 255;
  int c = threadIdx.x;
  float x = base[(size_t)idx * CH + c] + u[i * CH + c] + w[j * CH + c];
  __shared__ float red[128];
  red[c] = x;
  __syncthreads();
#pragma unroll
  for (int s = 64; s > 0; s >>= 1) {
    if (c < s) red[c] += red[c + s];
    __syncthreads();
  }
  float mean = red[0] * (1.f / 128.f);
  __syncthreads();
  float d = x - mean;
  red[c] = d * d;
  __syncthreads();
#pragma unroll
  for (int s = 64; s > 0; s >>= 1) {
    if (c < s) red[c] += red[c + s];
    __syncthreads();
  }
  float var = red[0] * (1.f / 128.f);
  float y = d * rsqrtf(var + LN_EPS) * g[c] + b[c];
  dstF[(size_t)idx * CH + c] = y;
  dstB[(size_t)idx * CH + c] = (bf16)y;
}

// --- templated WMMA GEMM: D = A[MxK] * Bt[NnxK]^T (+epilogue) ---------------
// Block: 8 waves = 2 M-tiles x 4 N-quads -> 32M x 128N. Each wave: 16x32
// (two accumulators, A fragment reused). Whole 32xK A tile async-staged to
// LDS once; inner loop is fully unrolled ds_load+WMMA (no barriers).
// EPI 0: bias->bf16   EPI 1: bias+relu->bf16   EPI 4: qkv (bias, Q cols *= sc)
// EPI 2: bias+res+LN  EPI 3: bias+relu+res+LN  (EPI 2/3 require Nn==128)
template <int EPI, int K>
__global__ __launch_bounds__(256) void k_gemm(
    const bf16* __restrict__ A, const bf16* __restrict__ Bt,
    const float* __restrict__ bias, const float* __restrict__ res,
    const float* __restrict__ g, const float* __restrict__ bvec,
    float* __restrict__ dstF, bf16* __restrict__ dstB, int Nn) {
  const int wave = threadIdx.x >> 5, lane = threadIdx.x & 31;
  const int mt = wave >> 2;  // 0..1
  const int nq = wave & 3;   // 0..3
  const int m0 = blockIdx.x * 32;
  const int n0 = blockIdx.y * 128;
  __shared__ __align__(16) bf16 as_a[32][K];  // full A tile: 32 rows x K
  const unsigned int a_base = lds_off_of(&as_a[0][0]);

  // async-stage A[m0..m0+31][0..K-1] -> LDS (16B chunks, 4*K total)
  const int cpr = K / 8;  // 16B chunks per row
#pragma unroll
  for (int c = threadIdx.x; c < 4 * K; c += 256) {
    async_copy16(a_base + c * 16, A + (size_t)(m0 + c / cpr) * K + (c % cpr) * 8);
  }

  const bf16* brow0 = Bt + (size_t)(n0 + nq * 32 + (lane & 15)) * K;
  const bf16* brow1 = brow0 + (size_t)16 * K;
  __builtin_prefetch(brow0, 0, 1);  // gfx1250 global_prefetch_b8
  __builtin_prefetch(brow1, 0, 1);

  wait_async0();
  __syncthreads();

  v8f acc0, acc1;
#pragma unroll
  for (int r = 0; r < 8; ++r) { acc0[r] = 0.f; acc1[r] = 0.f; }
#pragma unroll
  for (int k0 = 0; k0 < K; k0 += 32) {
    v16bf af = load_frag(&as_a[mt * 16 + (lane & 15)][k0], lane);
    v16bf bf0 = load_frag(brow0 + k0, lane);
    v16bf bf1 = load_frag(brow1 + k0, lane);
    acc0 = wmma_bf16(af, bf0, acc0);
    acc1 = wmma_bf16(af, bf1, acc1);
  }

  const int n = lane & 15;
  const int mh = (lane >> 4) << 3;
  const int nbase = n0 + nq * 32;
  const int mrow0 = m0 + mt * 16;
  if constexpr (EPI <= 1 || EPI == 4) {
    float b0 = bias[nbase + n], b1v = bias[nbase + 16 + n];
    const float sc0 = (EPI == 4 && (nbase + n) < CH) ? QK_SC : 1.f;
    const float sc1 = (EPI == 4 && (nbase + 16 + n) < CH) ? QK_SC : 1.f;
#pragma unroll
    for (int r = 0; r < 8; ++r) {
      float v0 = (acc0[r] + b0) * sc0;
      float v1 = (acc1[r] + b1v) * sc1;
      if constexpr (EPI == 1) {
        v0 = v0 > 0.f ? v0 : 0.f;
        v1 = v1 > 0.f ? v1 : 0.f;
      }
      dstB[(size_t)(mrow0 + mh + r) * Nn + nbase + n] = (bf16)v0;
      dstB[(size_t)(mrow0 + mh + r) * Nn + nbase + 16 + n] = (bf16)v1;
    }
  } else {
    __shared__ float tile[32][128];
    float b0 = bias[nbase + n], b1v = bias[nbase + 16 + n];
#pragma unroll
    for (int r = 0; r < 8; ++r) {
      float v0 = acc0[r] + b0;
      float v1 = acc1[r] + b1v;
      if constexpr (EPI == 3) {
        v0 = v0 > 0.f ? v0 : 0.f;
        v1 = v1 > 0.f ? v1 : 0.f;
      }
      int lrow = mt * 16 + mh + r;
      v0 += res[(size_t)(m0 + lrow) * 128 + nbase + n];
      v1 += res[(size_t)(m0 + lrow) * 128 + nbase + 16 + n];
      tile[lrow][nbase + n] = v0;
      tile[lrow][nbase + 16 + n] = v1;
    }
    __syncthreads();
    __shared__ float rs[32][8], rs2[32][8];
    int row = threadIdx.x >> 3, sub = threadIdx.x & 7;  // 8 threads per row
    float s = 0.f, s2 = 0.f;
#pragma unroll
    for (int c = sub * 16; c < sub * 16 + 16; ++c) {
      float x = tile[row][c];
      s += x;
      s2 += x * x;
    }
    rs[row][sub] = s;
    rs2[row][sub] = s2;
    __syncthreads();
    float ts = 0.f, ts2 = 0.f;
#pragma unroll
    for (int t = 0; t < 8; ++t) { ts += rs[row][t]; ts2 += rs2[row][t]; }
    float mean = ts * (1.f / 128.f);
    float var = ts2 * (1.f / 128.f) - mean * mean;
    float rstd = rsqrtf(var + LN_EPS);
#pragma unroll
    for (int c = sub * 16; c < sub * 16 + 16; ++c) {
      float x = (tile[row][c] - mean) * rstd * g[c] + bvec[c];
      size_t off = (size_t)(m0 + row) * 128 + c;
      dstF[off] = x;
      if (dstB) dstB[off] = (bf16)x;
    }
  }
}

// --- flash-style triangle attention -----------------------------------------
// grid(2, H, N); 8 waves/block, each wave owns 16 query rows of pair-row i.
// Q is pre-scaled by 1/sqrt(d) in the qkv GEMM epilogue. K chunk async-staged
// to LDS; V^T staged manually (transpose); P transposed through per-wave LDS
// back into WMMA A-layout. Softmax row-sums via WMMA against an all-ones
// fragment; row-max via DPP16 (no DS-pipe shuffles).
__global__ __launch_bounds__(256) void k_flash_attn(
    const bf16* __restrict__ qkv, const float* __restrict__ sbias,
    bf16* __restrict__ attn) {
  const int LD = 3 * CH;
  const int wave = threadIdx.x >> 5, lane = threadIdx.x & 31;
  const int h = blockIdx.y, i = blockIdx.z;
  const int j0 = blockIdx.x * 128 + wave * 16;
  const size_t base = (size_t)i * NSEQ * LD;

  __shared__ __align__(16) bf16 kt[32][32];      // K chunk (row-major, async)
  __shared__ __align__(16) bf16 vt[32][32];      // V^T chunk (manual transpose)
  __shared__ __align__(16) bf16 pst[8][16][32];  // per-wave P transpose stage
  const unsigned int k_base = lds_off_of(&kt[0][0]);

  const bf16* qrow = qkv + base + (size_t)(j0 + (lane & 15)) * LD + h * DHEAD;
  v16bf qf = load_frag(qrow, lane);

  v16bf ones;  // all-ones B fragment for row-sum WMMA
#pragma unroll
  for (int e = 0; e < 16; ++e) ones[e] = (bf16)1.0f;

  v8f o0, o1;
  float Mx[8], Lx[8], sold[8];
#pragma unroll
  for (int r = 0; r < 8; ++r) { o0[r] = 0.f; o1[r] = 0.f; Mx[r] = -INFINITY; Lx[r] = 0.f; }
  const int mh = (lane >> 4) << 3;
  const int n = lane & 15;
  const int skey = threadIdx.x >> 2;  // K staging (threads 0..127)
  const int spart = threadIdx.x & 3;

  for (int kc = 0; kc < 8; ++kc) {
    const int k0 = kc * 32;
    __syncthreads();
    if (threadIdx.x < 128) {  // async K[k0+key][h*32..+31] -> kt[key][.]
      async_copy16(k_base + skey * 64 + spart * 16,
                   qkv + base + (size_t)(k0 + skey) * LD + CH + h * DHEAD + spart * 8);
    }
    {  // manual V^T staging: vt[d][k] = V[k0+k][d]
      int kk = threadIdx.x >> 3;
      int d0 = (threadIdx.x & 7) * 4;
      const bf16* vp = qkv + base + (size_t)(k0 + kk) * LD + 2 * CH + h * DHEAD + d0;
      bf16 a0 = vp[0], a1 = vp[1], a2 = vp[2], a3 = vp[3];
      vt[d0 + 0][kk] = a0; vt[d0 + 1][kk] = a1;
      vt[d0 + 2][kk] = a2; vt[d0 + 3][kk] = a3;
    }
    wait_async0();
    __syncthreads();

    // S = (Q*sc) K^T : one WMMA (K=d=32) per 16x16 score tile
    v16bf kf0 = load_frag(&kt[n][0], lane);
    v16bf kf1 = load_frag(&kt[16 + n][0], lane);
    v8f z;
#pragma unroll
    for (int r = 0; r < 8; ++r) z[r] = 0.f;
    v8f s0 = wmma_bf16(qf, kf0, z);
    v8f s1 = wmma_bf16(qf, kf1, z);

    // online softmax: row max via DPP16, P staged to LDS
#pragma unroll
    for (int r = 0; r < 8; ++r) {
      int j = j0 + mh + r;
      float v0 = s0[r] + sbias[j * NSEQ + k0 + n];
      float v1 = s1[r] + sbias[j * NSEQ + k0 + 16 + n];
      float mx = row_max16(fmaxf(v0, v1));
      float Mn = fmaxf(Mx[r], mx);
      sold[r] = expf(Mx[r] - Mn);
      float p0 = expf(v0 - Mn), p1 = expf(v1 - Mn);
      o0[r] *= sold[r];
      o1[r] *= sold[r];
      Mx[r] = Mn;
      int m = mh + r;
      pst[wave][m][n] = (bf16)p0;
      pst[wave][m][16 + n] = (bf16)p1;
    }
    __syncthreads();

    // row sums of P via WMMA against ones; then O += P V
    v16bf pf = load_frag(&pst[wave][n][0], lane);
    v8f ssum = wmma_bf16(pf, ones, z);
#pragma unroll
    for (int r = 0; r < 8; ++r) Lx[r] = Lx[r] * sold[r] + ssum[r];
    v16bf vf0 = load_frag(&vt[n][0], lane);
    v16bf vf1 = load_frag(&vt[16 + n][0], lane);
    o0 = wmma_bf16(pf, vf0, o0);
    o1 = wmma_bf16(pf, vf1, o1);
  }

  const size_t ob = (size_t)i * NSEQ * CH;
#pragma unroll
  for (int r = 0; r < 8; ++r) {
    float inv = Lx[r] > 0.f ? 1.f / Lx[r] : 0.f;
    int j = j0 + mh + r;
    attn[ob + (size_t)j * CH + h * DHEAD + n] = (bf16)(o0[r] * inv);
    attn[ob + (size_t)j * CH + h * DHEAD + 16 + n] = (bf16)(o1[r] * inv);
  }
}

// ---------------------------------------------------------------------------
extern "C" void kernel_launch(void* const* d_in, const int* in_sizes, int n_in,
                              void* d_out, int out_size, void* d_ws, size_t ws_size,
                              hipStream_t stream) {
  (void)in_sizes; (void)n_in; (void)out_size; (void)ws_size;
  const float* pair      = (const float*)d_in[0];
  const float* single    = (const float*)d_in[1];
  const unsigned char* mask = (const unsigned char*)d_in[2];  // bool array
  const float* dist      = (const float*)d_in[3];
  const float* row_Wqkv  = (const float*)d_in[4];
  const float* row_bqkv  = (const float*)d_in[5];
  const float* row_Wo    = (const float*)d_in[6];
  const float* row_bo    = (const float*)d_in[7];
  const float* row_dsc   = (const float*)d_in[8];
  const float* col_Wqkv  = (const float*)d_in[9];
  const float* col_bqkv  = (const float*)d_in[10];
  const float* col_Wo    = (const float*)d_in[11];
  const float* col_bo    = (const float*)d_in[12];
  const float* col_dsc   = (const float*)d_in[13];
  const float* pnr_g     = (const float*)d_in[14];
  const float* pnr_b     = (const float*)d_in[15];
  const float* pnc_g     = (const float*)d_in[16];
  const float* pnc_b     = (const float*)d_in[17];
  const float* sn_g      = (const float*)d_in[18];
  const float* sn_b      = (const float*)d_in[19];
  const float* W1        = (const float*)d_in[20];
  const float* b1        = (const float*)d_in[21];
  const float* W2        = (const float*)d_in[22];
  const float* b2        = (const float*)d_in[23];

  float* outPr = (float*)d_out;                // [N,N,C] f32
  float* outSr = outPr + (size_t)NN_TOK * CH;  // [N,C] f32

  // workspace carve-up
  char* p = (char*)d_ws;
  auto alloc = [&](size_t bytes) {
    char* r = p;
    p += (bytes + 255) & ~(size_t)255;
    return r;
  };
  bf16* prb     = (bf16*)alloc((size_t)NN_TOK * CH * 2);      // pair (bf16)
  bf16* qkvb    = (bf16*)alloc((size_t)NN_TOK * 3 * CH * 2);  // qkv
  bf16* attnb   = (bf16*)alloc((size_t)NN_TOK * CH * 2);      // attn out
  float* biasR  = (float*)alloc((size_t)NN_TOK * 4);
  float* biasC  = (float*)alloc((size_t)NN_TOK * 4);
  float* srow   = (float*)alloc((size_t)NSEQ * CH * 4);
  float* scol   = (float*)alloc((size_t)NSEQ * CH * 4);
  bf16* wqkvrT  = (bf16*)alloc((size_t)3 * CH * CH * 2);
  bf16* wqkvcT  = (bf16*)alloc((size_t)3 * CH * CH * 2);
  bf16* worT    = (bf16*)alloc((size_t)CH * CH * 2);
  bf16* wocT    = (bf16*)alloc((size_t)CH * CH * 2);
  bf16* w1T     = (bf16*)alloc((size_t)CH * 2 * CH * 2);
  bf16* w2T     = (bf16*)alloc((size_t)2 * CH * CH * 2);
  bf16* singleb = (bf16*)alloc((size_t)NSEQ * CH * 2);
  bf16* h1b     = (bf16*)alloc((size_t)NSEQ * 2 * CH * 2);

  // weight prep + attention bias prep
  k_tconv<<<(CH * 384 + 255) / 256, 256, 0, stream>>>(row_Wqkv, wqkvrT, CH, 384);
  k_tconv<<<(CH * 384 + 255) / 256, 256, 0, stream>>>(col_Wqkv, wqkvcT, CH, 384);
  k_tconv<<<(CH * CH + 255) / 256, 256, 0, stream>>>(row_Wo, worT, CH, CH);
  k_tconv<<<(CH * CH + 255) / 256, 256, 0, stream>>>(col_Wo, wocT, CH, CH);
  k_tconv<<<(CH * 256 + 255) / 256, 256, 0, stream>>>(W1, w1T, CH, 256);
  k_tconv<<<(256 * CH + 255) / 256, 256, 0, stream>>>(W2, w2T, 256, CH);
  k_conv<<<(NSEQ * CH + 255) / 256, 256, 0, stream>>>(single, singleb, NSEQ * CH);
  k_bias<<<NN_TOK / 256, 256, 0, stream>>>(dist, mask, row_dsc, biasR, 0);
  k_bias<<<NN_TOK / 256, 256, 0, stream>>>(dist, mask, col_dsc, biasC, 1);

  // --- pr = LN(pair + tri_update_out(pair, single)) ---
  k_prefix_row<<<NSEQ, 128, 0, stream>>>(pair, srow);
  k_add_ln<<<NN_TOK, 128, 0, stream>>>(pair, srow, single, pnr_g, pnr_b, outPr, prb);

  // --- pr = LN(pr + row tri_attn(pr)) ---
  k_gemm<4, 128><<<dim3(NN_TOK / 32, 3), 256, 0, stream>>>(
      prb, wqkvrT, row_bqkv, nullptr, nullptr, nullptr, nullptr, qkvb, 384);
  k_flash_attn<<<dim3(2, HEADS, NSEQ), 256, 0, stream>>>(qkvb, biasR, attnb);
  k_gemm<2, 128><<<dim3(NN_TOK / 32, 1), 256, 0, stream>>>(
      attnb, worT, row_bo, outPr, pnr_g, pnr_b, outPr, prb, CH);

  // --- pr = LN(pr + tri_update_in(pr, single)) ---
  k_prefix_col<<<NSEQ, 128, 0, stream>>>(outPr, scol);
  k_add_ln<<<NN_TOK, 128, 0, stream>>>(outPr, single, scol, pnc_g, pnc_b, outPr, prb);

  // --- pr = LN(pr + col tri_attn(pr, mask^T)) ---
  k_gemm<4, 128><<<dim3(NN_TOK / 32, 3), 256, 0, stream>>>(
      prb, wqkvcT, col_bqkv, nullptr, nullptr, nullptr, nullptr, qkvb, 384);
  k_flash_attn<<<dim3(2, HEADS, NSEQ), 256, 0, stream>>>(qkvb, biasC, attnb);
  k_gemm<2, 128><<<dim3(NN_TOK / 32, 1), 256, 0, stream>>>(
      attnb, wocT, col_bo, outPr, pnc_g, pnc_b, outPr, prb, CH);

  // --- sr = LN(single + relu(relu(single@W1+b1)@W2+b2)) ---
  k_gemm<1, 128><<<dim3(NSEQ / 32, 2), 256, 0, stream>>>(
      singleb, w1T, b1, nullptr, nullptr, nullptr, nullptr, h1b, 256);
  k_gemm<3, 256><<<dim3(NSEQ / 32, 1), 256, 0, stream>>>(
      h1b, w2T, b2, single, sn_g, sn_b, outSr, nullptr, CH);
}
